// convKxK_13657996001488
// MI455X (gfx1250) — matrix operand order repaired and checked by reading, not compiled, not run on hardware
//
#include <hip/hip_runtime.h>
#include <hip/hip_bf16.h>

// ---------------------------------------------------------------------------
// DeepGCN block on MI455X (gfx1250, wave32, WMMA 16x16x32 f16):
//   stage A: EdgeConv   feat[1M x 128] @ w_edge^T[128 x 64]  (+BN stats, 2-pass)
//   stage B: WeightedConv1D  [B*N x 576] @ w_conv'[576 x 128]
// Both GEMMs run on v_wmma_f32_16x16x32_f16 (f16 in, f32 accumulate).
// ---------------------------------------------------------------------------

typedef __attribute__((ext_vector_type(16))) _Float16 v16h;
typedef __attribute__((ext_vector_type(8)))  float    v8f;

#define B_SZ   4
#define N_SZ   16384
#define K_NN   16
#define C_INP  64
#define C_OUTP 128
#define KSIZE  9
#define PADW   4
#define CNT_F  (4.0f * 16384.0f * 16.0f)     /* B*N*K samples for BN stats */
#define EPS_BN 1e-5f
#define NEG_SL 0.2f
#define INV_SIG2 2500.0f                     /* 1/sigma^2, sigma=0.02 */

// LDS row pitches in halves (padded to rotate banks across rows)
#define APITCH 136   /* 128 + 8 */
#define BPITCH 136
#define WPITCH 72    /* 64 + 8 */

// ---------------------------------------------------------------------------
// WMMA fragment loaders from LDS per the CDNA5 ISA 16-bit layouts.
// A (16x32, MxK): lane<16 -> M=lane, K = {0..7,16..23}; lane>=16 -> M=lane-16,
//                 K = {8..15,24..31}. K comes in consecutive pairs -> dwords.
// B (32x16, KxN): stored as rows of N (one LDS row per output column), same
//                 K order as A: lane<16 -> K = {0..7,16..23};
//                 lane>=16 -> K = {8..15,24..31}.
// ---------------------------------------------------------------------------
__device__ inline v16h load_a_frag(const _Float16* row, int kc, int lane) {
  const unsigned int* r32 = (const unsigned int*)row;
  int base = kc * 16 + ((lane < 16) ? 0 : 4);
  union { unsigned int u[8]; v16h v; } f;
#pragma unroll
  for (int p = 0; p < 8; ++p)
    f.u[p] = r32[base + (p < 4 ? p : p + 4)];
  return f.v;
}

__device__ inline v16h load_b_frag(const _Float16* row, int kc, int lane) {
  const unsigned int* r32 = (const unsigned int*)row;
  int base = kc * 16 + ((lane < 16) ? 0 : 4);
  union { unsigned int u[8]; v16h v; } f;
#pragma unroll
  for (int p = 0; p < 8; ++p)
    f.u[p] = r32[base + (p < 4 ? p : p + 4)];
  return f.v;
}

__device__ inline unsigned int pack2h(float a, float b) {
  union { _Float16 h[2]; unsigned int u; } p;
  p.h[0] = (_Float16)a;
  p.h[1] = (_Float16)b;
  return p.u;
}

// ---------------------------------------------------------------------------
// 0) one-time prep: zero BN stats; convert w_edge -> f16; rearrange w_conv
//    to tap-major f16 [j][o][c] so per-tap LDS staging is pure dword copies.
//    grid: 320 x 256 = 81920 threads = 8192 + 73728 elements.
// ---------------------------------------------------------------------------
__global__ __launch_bounds__(256)
void prep_kernel(const float* __restrict__ w_edge,   // [64][128]
                 const float* __restrict__ w_conv,   // [128][64][9]
                 _Float16* __restrict__ we16,        // [64][128]
                 _Float16* __restrict__ wc16,        // [9][128][64]
                 float* __restrict__ gstats) {
  const int gid = blockIdx.x * 256 + threadIdx.x;
  if (blockIdx.x == 0 && threadIdx.x < 128) gstats[threadIdx.x] = 0.f;
  if (gid < 64 * 128) {
    we16[gid] = (_Float16)w_edge[gid];
  } else {
    int t = gid - 64 * 128;                  // < 9*128*64 = 73728
    int j = t >> 13;                         // / 8192
    int rem = t & 8191;
    int o = rem >> 6, c = rem & 63;
    wc16[t] = (_Float16)w_conv[(o * 64 + c) * KSIZE + j];
  }
}

// ---------------------------------------------------------------------------
// 1) transpose x [B,64,N] -> xt [B,N,64] so edge gathers read 256B rows
// ---------------------------------------------------------------------------
__global__ __launch_bounds__(256)
void transpose_kernel(const float* __restrict__ x, float* __restrict__ xt) {
  __shared__ float t[64][65];
  const int b = blockIdx.y, n0 = blockIdx.x * 64, tid = threadIdx.x;
  for (int i = tid; i < 64 * 64; i += 256) {
    int c = i >> 6, nn = i & 63;
    t[c][nn] = x[((long)b * C_INP + c) * N_SZ + n0 + nn];   // coalesced in n
  }
  __syncthreads();
  for (int i = tid; i < 64 * 64; i += 256) {
    int nn = i >> 6, c = i & 63;
    xt[((long)b * N_SZ + n0 + nn) * C_INP + c] = t[c][nn];  // coalesced in c
  }
}

// ---------------------------------------------------------------------------
// 2/3) EdgeConv GEMM. One wave = one point (M = its 16 neighbors).
//   PASS 0: accumulate per-channel sum / sumsq of h into gstats (atomics).
//   PASS 1: recompute h, apply BN + LeakyReLU, max over K, store transposed
//           h_t [B,N,64].
// ---------------------------------------------------------------------------
template <int PASS>
__global__ __launch_bounds__(256)
void edge_gemm_kernel(const float* __restrict__ xt,        // [B][N][64]
                      const int*   __restrict__ eidx,      // [2][B][N][K]
                      const _Float16* __restrict__ we16,   // [64][128] f16
                      const float* __restrict__ gamma,
                      const float* __restrict__ beta,
                      float* __restrict__ gstats,          // sum[64]|sumsq[64]
                      float* __restrict__ h_t) {           // [B][N][64]
  __shared__ _Float16 sB[64 * BPITCH];        // w_edge f16, row = out ch
  __shared__ _Float16 sA[8][16 * APITCH];     // per-wave A tiles
  __shared__ float red0[64];                  // pass0: sum   | pass1: scale
  __shared__ float red1[64];                  // pass0: sumsq | pass1: shift

  const int tid = threadIdx.x;
  const int wave = tid >> 5;
  const int lane = tid & 31;

  // stage w_edge: pure dword copies from the f16 shadow
  {
    const unsigned int* w32 = (const unsigned int*)we16;
    for (int i = tid; i < 64 * 64; i += 256) {          // 4096 dwords
      int o = i >> 6, cp = i & 63;
      *(unsigned int*)&sB[o * BPITCH + 2 * cp] = w32[i];
    }
  }
  if (PASS == 0) {
    if (tid < 64) { red0[tid] = 0.f; red1[tid] = 0.f; }
  } else {
    if (tid < 64) {
      float mean = gstats[tid] * (1.0f / CNT_F);
      float var  = gstats[64 + tid] * (1.0f / CNT_F) - mean * mean;
      float sc   = gamma[tid] * rsqrtf(var + EPS_BN);
      red0[tid] = sc;
      red1[tid] = beta[tid] - mean * sc;
    }
  }
  __syncthreads();

  const int pid = blockIdx.x * 8 + wave;       // point id in [0, B*N)
  const int b = pid >> 14;                     // N = 2^14
  const int n = pid & (N_SZ - 1);

  // lane-parallel preload of all 32 edge indices, then shfl-broadcast per k
  const long ebase = ((long)b * N_SZ + n) * K_NN;
  long eoff = (lane < 16) ? (ebase + lane)
                          : ((long)B_SZ * N_SZ * K_NN + ebase + (lane - 16));
  const int ev = eidx[eoff];                   // lanes 0-15: e0(j), 16-31: e1(i)

  // ---- gather & build feat tile: [16 k-rows x 128] = [x_i | x_j - x_i] ----
  // lane owns channel pair (2*lane, 2*lane+1): float2 gather + packed b32 DS.
  _Float16* A = sA[wave];
#pragma unroll
  for (int k = 0; k < K_NN; ++k) {
    int jdx = __shfl(ev, k);
    int idx = __shfl(ev, 16 + k);
    const float2 xi2 = *(const float2*)(xt + ((long)b * N_SZ + idx) * C_INP + 2 * lane);
    const float2 xj2 = *(const float2*)(xt + ((long)b * N_SZ + jdx) * C_INP + 2 * lane);
    *(unsigned int*)&A[k * APITCH + 2 * lane]      = pack2h(xi2.x, xi2.y);
    *(unsigned int*)&A[k * APITCH + 64 + 2 * lane] = pack2h(xj2.x - xi2.x, xj2.y - xi2.y);
  }
  // same-wave LDS RAW: DS ops are in-order per wave; compiler inserts waits.

  // ---- h[16 x 64] = A[16 x 128] * B[128 x 64] : 16 WMMAs ----
  const int m = lane & 15;
  const _Float16* arow = A + m * APITCH;
  v8f acc[4];
#pragma unroll
  for (int nt = 0; nt < 4; ++nt)
    acc[nt] = (v8f){0.f, 0.f, 0.f, 0.f, 0.f, 0.f, 0.f, 0.f};
#pragma unroll
  for (int kc = 0; kc < 4; ++kc) {
    v16h af = load_a_frag(arow, kc, lane);
#pragma unroll
    for (int nt = 0; nt < 4; ++nt) {
      const _Float16* brow = sB + (nt * 16 + m) * BPITCH;
      v16h bf = load_b_frag(brow, kc, lane);
      acc[nt] = __builtin_amdgcn_wmma_f32_16x16x32_f16(
          false, af, false, bf, (short)0, acc[nt], false, false);
    }
  }

  if (PASS == 0) {
    // C layout: VGPR r -> M (=k), lane&15 -> N (=channel). Reduce over M.
#pragma unroll
    for (int nt = 0; nt < 4; ++nt) {
      float s = 0.f, q = 0.f;
#pragma unroll
      for (int r = 0; r < 8; ++r) { float v = acc[nt][r]; s += v; q += v * v; }
      int ch = nt * 16 + m;
      atomicAdd(&red0[ch], s);
      atomicAdd(&red1[ch], q);
    }
    __syncthreads();
    if (tid < 64) {
      atomicAdd(&gstats[tid], red0[tid]);
      atomicAdd(&gstats[64 + tid], red1[tid]);
    }
  } else {
#pragma unroll
    for (int nt = 0; nt < 4; ++nt) {
      int ch = nt * 16 + m;
      float sc = red0[ch], sh = red1[ch];
      float mx = -1e30f;
#pragma unroll
      for (int r = 0; r < 8; ++r) {
        float v = acc[nt][r] * sc + sh;
        v = (v >= 0.f) ? v : NEG_SL * v;          // leaky_relu(0.2)
        mx = fmaxf(mx, v);
      }
      // lane L holds M=0..7, lane L+16 holds M=8..15 for the same channel:
      mx = fmaxf(mx, __shfl_xor(mx, 16));         // max over all K=16
      if (lane < 16)
        h_t[((long)b * N_SZ + n) * C_INP + ch] = mx;
    }
  }
}

// ---------------------------------------------------------------------------
// 4) WeightedConv1D. One wave = 16 consecutive points; loop 9 taps, each tap
//    stages wc16[j] (f16, 16KB) in LDS and folds exp(-d2/sigma^2) into the
//    A operand. 144 WMMAs per wave. Gaussian weights computed once per tap
//    by lanes 0-15 (one point each) and shfl-broadcast during the A build.
// ---------------------------------------------------------------------------
__global__ __launch_bounds__(256)
void wconv_kernel(const float* __restrict__ h_t,       // [B][N][64]
                  const float* __restrict__ coords,    // [B][3][N]
                  const _Float16* __restrict__ wc16,   // [9][128][64] f16
                  const float* __restrict__ b_conv,    // [128]
                  float* __restrict__ out) {           // [B][128][N]
  __shared__ _Float16 sB[128 * WPITCH];
  __shared__ _Float16 sA[8][16 * WPITCH];

  const int tid = threadIdx.x, wave = tid >> 5, lane = tid & 31;
  const int b = blockIdx.y;
  const int n0w = blockIdx.x * 128 + wave * 16;
  const int m = lane & 15;
  _Float16* A = sA[wave];
  const float* cb = coords + (long)b * 3 * N_SZ;

  // center coords for this lane's point (m = lane&15; halves duplicate work)
  const int nc = n0w + m;
  const float cx = cb[nc], cy = cb[N_SZ + nc], cz = cb[2 * N_SZ + nc];

  v8f acc[8];
#pragma unroll
  for (int nt = 0; nt < 8; ++nt)
    acc[nt] = (v8f){0.f, 0.f, 0.f, 0.f, 0.f, 0.f, 0.f, 0.f};

  for (int j = 0; j < KSIZE; ++j) {
    {   // stage wc16[j] as Bt[o][c]: pure dword copies
      const unsigned int* w32 = (const unsigned int*)(wc16 + (long)j * 128 * 64);
      for (int i = tid; i < 128 * 32; i += 256) {       // 4096 dwords
        int o = i >> 5, cp = i & 31;
        *(unsigned int*)&sB[o * WPITCH + 2 * cp] = w32[i];
      }
    }
    __syncthreads();

    // per-tap Gaussian weight for this lane's point (coalesced coord loads)
    float w = 0.f;
    {
      int np = nc + j - PADW;
      if (np >= 0 && np < N_SZ) {
        float dx = cb[np] - cx;
        float dy = cb[N_SZ + np] - cy;
        float dz = cb[2 * N_SZ + np] - cz;
        w = __expf(-(dx * dx + dy * dy + dz * dz) * INV_SIG2);
      }
    }

    // build A tile: lane owns channel pair (2*lane, 2*lane+1)
#pragma unroll
    for (int mm = 0; mm < 16; ++mm) {
      float wm = __shfl(w, mm);                         // weight of point mm
      int np = n0w + mm + j - PADW;
      unsigned int pk = 0u;                             // zero-padded window
      if (np >= 0 && np < N_SZ) {
        const float2 h2 = *(const float2*)(h_t + ((long)b * N_SZ + np) * C_INP + 2 * lane);
        pk = pack2h(h2.x * wm, h2.y * wm);
      }
      *(unsigned int*)&A[mm * WPITCH + 2 * lane] = pk;
    }

    const _Float16* arow = A + m * WPITCH;
#pragma unroll
    for (int kc = 0; kc < 2; ++kc) {
      v16h af = load_a_frag(arow, kc, lane);
#pragma unroll
      for (int nt = 0; nt < 8; ++nt) {
        const _Float16* brow = sB + (nt * 16 + m) * WPITCH;
        v16h bf = load_b_frag(brow, kc, lane);
        acc[nt] = __builtin_amdgcn_wmma_f32_16x16x32_f16(
            false, af, false, bf, (short)0, acc[nt], false, false);
      }
    }
    __syncthreads();                                    // sB reload next tap
  }

  // epilogue: bias + store; each lane writes 8 consecutive n for its channel
#pragma unroll
  for (int nt = 0; nt < 8; ++nt) {
    int o = nt * 16 + m;
    float bias = b_conv[o];
    int nbase = n0w + ((lane < 16) ? 0 : 8);
    float* op = out + ((long)b * C_OUTP + o) * N_SZ + nbase;
#pragma unroll
    for (int r = 0; r < 8; ++r) op[r] = acc[nt][r] + bias;
  }
}

// ---------------------------------------------------------------------------
extern "C" void kernel_launch(void* const* d_in, const int* in_sizes, int n_in,
                              void* d_out, int out_size, void* d_ws,
                              size_t ws_size, hipStream_t stream) {
  const float* x      = (const float*)d_in[0];
  const float* coords = (const float*)d_in[1];
  const int*   eidx   = (const int*)d_in[2];
  const float* w_edge = (const float*)d_in[3];
  const float* gamma  = (const float*)d_in[4];
  const float* beta   = (const float*)d_in[5];
  const float* w_conv = (const float*)d_in[6];
  const float* b_conv = (const float*)d_in[7];
  float* out = (float*)d_out;

  // workspace: [stats 1KB | we16 16KB | wc16 144KB | xt 16.8MB | h_t 16.8MB]
  char* ws = (char*)d_ws;
  float*    gstats = (float*)ws;
  _Float16* we16   = (_Float16*)(ws + 1024);
  _Float16* wc16   = (_Float16*)(ws + 1024 + 64 * 128 * sizeof(_Float16));
  char*     p      = ws + 1024 + 64 * 128 * sizeof(_Float16)
                        + (size_t)KSIZE * 128 * 64 * sizeof(_Float16);
  float*    xt     = (float*)p;
  float*    h_t    = (float*)(p + (size_t)B_SZ * N_SZ * C_INP * sizeof(float));

  prep_kernel<<<dim3(320), dim3(256), 0, stream>>>(w_edge, w_conv, we16, wc16, gstats);
  transpose_kernel<<<dim3(N_SZ / 64, B_SZ), dim3(256), 0, stream>>>(x, xt);
  edge_gemm_kernel<0><<<dim3(B_SZ * N_SZ / 8), dim3(256), 0, stream>>>(
      xt, eidx, we16, gamma, beta, gstats, h_t);
  edge_gemm_kernel<1><<<dim3(B_SZ * N_SZ / 8), dim3(256), 0, stream>>>(
      xt, eidx, we16, gamma, beta, gstats, h_t);
  wconv_kernel<<<dim3(N_SZ / 128, B_SZ), dim3(256), 0, stream>>>(
      h_t, coords, wc16, b_conv, out);
}
